// ZeusConv_25323127177751
// MI455X (gfx1250) — matrix operand order repaired and checked by reading, not compiled, not run on hardware
//
#include <hip/hip_runtime.h>
#include <hip/hip_bf16.h>
#include <stddef.h>

// ---------------------------------------------------------------------------
// Reference: B=32, C=768, HW=1024. Only the K,V chunks of W_in (rows 0..2C)
// are used. Per batch GEMM: [1536 x 768] x [768 x 1024], fused with per-channel
// cosine-similarity gate (K,V never touch global memory).
// ---------------------------------------------------------------------------
#define BATCH 32
#define CH    768
#define HW    1024

typedef __attribute__((ext_vector_type(16))) __bf16 v16bf;
typedef __attribute__((ext_vector_type(8)))  __bf16 v8bf;
typedef __attribute__((ext_vector_type(8)))  float  v8f;

__device__ __forceinline__ unsigned bf16bits(float f) {
    union { float f; unsigned u; } v; v.f = f;
    return (v.u + 0x7FFFu + ((v.u >> 16) & 1u)) >> 16;   // round-nearest-even
}
__device__ __forceinline__ __bf16 f2bf(float f) {
    union { unsigned short s; __bf16 b; } o; o.s = (unsigned short)bf16bits(f);
    return o.b;
}
// pack two floats -> packed bf16x2 dword
__device__ __forceinline__ unsigned pk2bf(float lo, float hi) {
    return bf16bits(lo) | (bf16bits(hi) << 16);
}

// ---------------------------------------------------------------------------
// Kernel 1: rinv[b,s] = rsqrt(mean_c X[b,c,s]^2 + 1e-6).
// One block per batch; each thread owns 4 consecutive positions (float4).
// ---------------------------------------------------------------------------
__global__ __launch_bounds__(256) void zeus_rms_kernel(
    const float* __restrict__ X, float* __restrict__ rinv)
{
    const int b  = blockIdx.x;
    const int s4 = threadIdx.x;                       // float4 index, 0..255
    const float4* xp = (const float4*)(X + (size_t)b * CH * HW) + s4;
    float a0 = 0.f, a1 = 0.f, a2 = 0.f, a3 = 0.f;
    #pragma unroll 4
    for (int c = 0; c < CH; ++c) {
        float4 v = xp[c * (HW / 4)];
        a0 = fmaf(v.x, v.x, a0); a1 = fmaf(v.y, v.y, a1);
        a2 = fmaf(v.z, v.z, a2); a3 = fmaf(v.w, v.w, a3);
    }
    const float s = 1.0f / (float)CH;
    float4 r;
    r.x = rsqrtf(a0 * s + 1e-6f); r.y = rsqrtf(a1 * s + 1e-6f);
    r.z = rsqrtf(a2 * s + 1e-6f); r.w = rsqrtf(a3 * s + 1e-6f);
    ((float4*)(rinv + (size_t)b * HW))[s4] = r;
}

// ---------------------------------------------------------------------------
// Kernel 2: fused GEMM + cosine gate.
// grid = (C/16 = 48, B = 32); block = 128 threads = 4 waves.
// Dynamic LDS layout (148224 B):
//   wk [16][768] bf16   K-proj weight rows            (24 KB)
//   wv [16][768] bf16   V-proj weight rows            (24 KB)
//   bt [64][768] bf16   full-depth B tile, [col][k]   (96 KB)
//   red[4][16][3] float reduction scratch
// K-loop is barrier-free: 6x ds_load_b128 + 2x v_wmma per step, unrolled.
// ---------------------------------------------------------------------------
__global__ __launch_bounds__(128) void zeus_gemm_attn_kernel(
    const float* __restrict__ X,     // [B, C, HW]
    const float* __restrict__ Wm,    // [3C, C]
    const float* __restrict__ rinv,  // [B, HW]
    float*       __restrict__ A)     // [B, C]
{
    extern __shared__ __align__(16) __bf16 smem[];
    __bf16* wk = smem;                       // 16*768
    __bf16* wv = wk + 16 * CH;               // 16*768
    __bf16* bt = wv + 16 * CH;               // 64*768
    float*  red = (float*)(bt + 64 * CH);    // [4][16][3]

    const int b    = blockIdx.y;
    const int c0   = blockIdx.x << 4;
    const int tid  = threadIdx.x;
    const int wave = tid >> 5;
    const int lane = tid & 31;
    const int half = lane >> 4;              // ISA 16-bit A/B layout half
    const int mrow = lane & 15;

    // ---- stage W rows as bf16 (coalesced over k) --------------------------
    for (int i = tid; i < 16 * CH; i += 128) {
        const int m = i / CH, k = i - m * CH;
        wk[i] = f2bf(Wm[(size_t)(c0 + m) * CH + k]);
        wv[i] = f2bf(Wm[(size_t)(CH + c0 + m) * CH + k]);
    }

    float akv[8] = {0,0,0,0,0,0,0,0};
    float akk[8] = {0,0,0,0,0,0,0,0};
    float avv[8] = {0,0,0,0,0,0,0,0};

    // staging decomposition: thread covers 4x4 (k x col) micro-tiles
    const int cq  = (tid & 15) << 2;         // col4: 0,4,..,60
    const int kq0 = tid >> 4;                // row-quad id 0..7

    for (int n0 = 0; n0 < HW; n0 += 64) {
        // ---- stage full-depth B tile: Xn[0:768, n0+0:64] -> bt[col][k] ----
        __syncthreads();                      // previous tile's reads done
        {
            const float4 rv = *(const float4*)&rinv[b * HW + n0 + cq];
            const float* xb = X + ((size_t)b * CH) * HW + n0 + cq;
            for (int kq = kq0; kq < CH / 4; kq += 8) {     // 24 micro-tiles
                const int k4 = kq << 2;
                float4 r0 = *(const float4*)&xb[(size_t)(k4 + 0) * HW];
                float4 r1 = *(const float4*)&xb[(size_t)(k4 + 1) * HW];
                float4 r2 = *(const float4*)&xb[(size_t)(k4 + 2) * HW];
                float4 r3 = *(const float4*)&xb[(size_t)(k4 + 3) * HW];
                r0.x *= rv.x; r0.y *= rv.y; r0.z *= rv.z; r0.w *= rv.w;
                r1.x *= rv.x; r1.y *= rv.y; r1.z *= rv.z; r1.w *= rv.w;
                r2.x *= rv.x; r2.y *= rv.y; r2.z *= rv.z; r2.w *= rv.w;
                r3.x *= rv.x; r3.y *= rv.y; r3.z *= rv.z; r3.w *= rv.w;
                // transpose 4x4 in registers; 4 consecutive k per col -> b64
                *(uint2*)&bt[(cq + 0) * CH + k4] = make_uint2(pk2bf(r0.x, r1.x), pk2bf(r2.x, r3.x));
                *(uint2*)&bt[(cq + 1) * CH + k4] = make_uint2(pk2bf(r0.y, r1.y), pk2bf(r2.y, r3.y));
                *(uint2*)&bt[(cq + 2) * CH + k4] = make_uint2(pk2bf(r0.z, r1.z), pk2bf(r2.z, r3.z));
                *(uint2*)&bt[(cq + 3) * CH + k4] = make_uint2(pk2bf(r0.w, r1.w), pk2bf(r2.w, r3.w));
            }
            if (n0 + 64 < HW)                 // warm L2/WGP$ for next tile
                __builtin_prefetch(&xb[(size_t)(kq0 * 96) * HW + 64], 0, 1);
        }
        __syncthreads();                      // tile visible to all waves

        // ---- barrier-free K loop: pure LDS b128 loads + WMMA --------------
        v8f ck = {}; v8f cv = {};
        const __bf16* wkp = wk + mrow * CH + (half << 3);
        const __bf16* wvp = wv + mrow * CH + (half << 3);
        const __bf16* btp = bt + ((wave << 4) + mrow) * CH + (half << 3);
        #pragma unroll 4
        for (int k0 = 0; k0 < CH; k0 += 32) {
            const v8bf ak_lo = *(const v8bf*)(wkp + k0);
            const v8bf ak_hi = *(const v8bf*)(wkp + k0 + 16);
            const v8bf av_lo = *(const v8bf*)(wvp + k0);
            const v8bf av_hi = *(const v8bf*)(wvp + k0 + 16);
            const v8bf b_lo  = *(const v8bf*)(btp + k0);
            const v8bf b_hi  = *(const v8bf*)(btp + k0 + 16);
            const v16bf a_k = __builtin_shufflevector(ak_lo, ak_hi,
                0,1,2,3,4,5,6,7,8,9,10,11,12,13,14,15);
            const v16bf a_v = __builtin_shufflevector(av_lo, av_hi,
                0,1,2,3,4,5,6,7,8,9,10,11,12,13,14,15);
            const v16bf bf  = __builtin_shufflevector(b_lo, b_hi,
                0,1,2,3,4,5,6,7,8,9,10,11,12,13,14,15);
            ck = __builtin_amdgcn_wmma_f32_16x16x32_bf16(
                     false, a_k, false, bf, (short)0, ck, false, false);
            cv = __builtin_amdgcn_wmma_f32_16x16x32_bf16(
                     false, a_v, false, bf, (short)0, cv, false, false);
        }

        // ---- gate epilogue: elementwise products on resident fragments ----
        #pragma unroll
        for (int r = 0; r < 8; ++r) {
            akv[r] = fmaf(ck[r], cv[r], akv[r]);
            akk[r] = fmaf(ck[r], ck[r], akk[r]);
            avv[r] = fmaf(cv[r], cv[r], avv[r]);
        }
    }

    // reduce over the 16 N-lanes inside each half (C/D layout: M=r+8*half, N=mrow)
    #pragma unroll
    for (int off = 1; off < 16; off <<= 1) {
        #pragma unroll
        for (int r = 0; r < 8; ++r) {
            akv[r] += __shfl_xor(akv[r], off, 32);
            akk[r] += __shfl_xor(akk[r], off, 32);
            avv[r] += __shfl_xor(avv[r], off, 32);
        }
    }
    if (mrow == 0) {
        #pragma unroll
        for (int r = 0; r < 8; ++r) {
            const int row = (half << 3) + r;
            red[(wave * 16 + row) * 3 + 0] = akv[r];
            red[(wave * 16 + row) * 3 + 1] = akk[r];
            red[(wave * 16 + row) * 3 + 2] = avv[r];
        }
    }
    __syncthreads();
    if (tid < 16) {
        float kv = 0.f, kk = 0.f, vv = 0.f;
        #pragma unroll
        for (int w = 0; w < 4; ++w) {
            kv += red[(w * 16 + tid) * 3 + 0];
            kk += red[(w * 16 + tid) * 3 + 1];
            vv += red[(w * 16 + tid) * 3 + 2];
        }
        const float cosv = kv / ((sqrtf(kk) + 1e-12f) * (sqrtf(vv) + 1e-12f));
        A[b * CH + c0 + tid] = 0.5f * cosv + 0.5f;
    }
}

// ---------------------------------------------------------------------------
// Kernel 3: out[b,c,:] = A[b,c] * X[b,c,:]; one block per (b,c) row.
// ---------------------------------------------------------------------------
__global__ __launch_bounds__(256) void zeus_scale_kernel(
    const float* __restrict__ X, const float* __restrict__ A,
    float* __restrict__ out)
{
    const int row = blockIdx.x;
    const float a = A[row];
    const float4* xi = (const float4*)(X + (size_t)row * HW);
    float4*       oi = (float4*)(out + (size_t)row * HW);
    float4 v = xi[threadIdx.x];
    v.x *= a; v.y *= a; v.z *= a; v.w *= a;
    oi[threadIdx.x] = v;
}

extern "C" void kernel_launch(void* const* d_in, const int* in_sizes, int n_in,
                              void* d_out, int out_size, void* d_ws, size_t ws_size,
                              hipStream_t stream) {
    (void)in_sizes; (void)n_in; (void)out_size; (void)ws_size;
    const float* X  = (const float*)d_in[0];   // [32,768,32,32]
    const float* Wm = (const float*)d_in[1];   // [2304,768]
    float* out  = (float*)d_out;

    float* rinv = (float*)d_ws;                // 32768 floats
    float* A    = rinv + BATCH * HW;           // 24576 floats

    // dynamic LDS: wk + wv + bt + red
    const size_t lds_bytes = (size_t)(16 * CH + 16 * CH + 64 * CH) * sizeof(__bf16)
                           + 4 * 16 * 3 * sizeof(float);

    zeus_rms_kernel<<<BATCH, 256, 0, stream>>>(X, rinv);
    zeus_gemm_attn_kernel<<<dim3(CH / 16, BATCH), 128, lds_bytes, stream>>>(X, Wm, rinv, A);
    zeus_scale_kernel<<<BATCH * CH, 256, 0, stream>>>(X, A, out);
}
